// MSDeformablePoints_74337293959224
// MI455X (gfx1250) — matrix operand order repaired and checked by reading, not compiled, not run on hardware
//
#include <hip/hip_runtime.h>
#include <hip/hip_bf16.h>

typedef __attribute__((ext_vector_type(16))) __bf16 v16bf;
typedef __attribute__((ext_vector_type(8)))  float  v8f;

union BFrag { uint4 q[2]; unsigned short u[16]; v16bf v; };

// ---------------------------------------------------------------- helpers

__device__ __forceinline__ unsigned short f2bf(float f) {
    unsigned u = __float_as_uint(f);
    unsigned r = u + 0x7FFFu + ((u >> 16) & 1u);   // round-to-nearest-even
    return (unsigned short)(r >> 16);
}

__device__ __forceinline__ float wave_sum(float v) {
#pragma unroll
    for (int m = 16; m > 0; m >>= 1) v += __shfl_xor(v, m, 32);
    return v;
}

#define NB        8
#define EMB       256
#define LTOT      21760   // 128*128 + 64*64 + 32*32 + 16*16

// ---------------------------------------------------------------- kernel 1
// Swizzle qw (4,256,256) f32 -> bf16 WMMA B-fragments.
// frag[level][ntile(16)][kstep(8)][lane(32)][t(16)]
//   n = ntile*16 + (lane&15);  k = kstep*32 + 8*(lane>>4) + (t<8 ? t : t+8);  B[k][n] = qw[n][k]
__global__ void prep_qw_kernel(const float* __restrict__ qw,
                               unsigned short* __restrict__ frag) {
    int i = blockIdx.x * blockDim.x + threadIdx.x;   // 0 .. 262143
    int t     =  i        & 15;
    int lane  = (i >> 4)  & 31;
    int ks    = (i >> 9)  & 7;
    int ntile = (i >> 12) & 15;
    int level =  i >> 16;
    int n = ntile * 16 + (lane & 15);
    int k = ks * 32 + 8 * (lane >> 4) + (t < 8 ? t : t + 8);
    frag[i] = f2bf(qw[(size_t)level * 65536 + (size_t)n * 256 + k]);
}

// ---------------------------------------------------------------- kernel 2
// Gather windowed x rows -> bf16 WMMA A-fragments (converted exactly once).
// afrag[level][mtile][kstep(8)][lane(32)][t(16)]
//   row = mtile*16 + (lane&15);  k = kstep*32 + (t<8 ? t+8g : 8+t+8g), g=lane>>4
// OOB(zero-pad) rows stored as zeros.
__global__ void __launch_bounds__(256)
aprep_kernel(const float* __restrict__ x, unsigned short* __restrict__ afrag,
             int K, int ST, int H, int lvlbase, unsigned long long aoff, int chunks) {
    int i = blockIdx.x * blockDim.x + threadIdx.x;   // one 16-elem chunk per thread
    if (i >= chunks) return;
    int lane  = i & 31;
    int ks    = (i >> 5) & 7;
    int mtile = i >> 8;
    int l15 = lane & 15, g = lane >> 4;
    int W = H, K2 = K * K, per_b = 64 * K2;

    int m    = mtile * 16 + l15;
    int b    = m / per_b;  int rem  = m - b * per_b;
    int p    = rem / K2;   int kidx = rem - p * K2;
    int py   = p >> 3, px = p & 7;
    int ky   = kidx / K,  kx = kidx - ky * K;
    int y    = py * ST + ky - (K >> 1);
    int xc   = px * ST + kx - (K >> 1);
    bool valid = ((unsigned)y < (unsigned)H) && ((unsigned)xc < (unsigned)W);

    BFrag A;
    if (valid) {
        const float* xrow = x + ((size_t)b * LTOT + lvlbase + (size_t)y * W + xc) * EMB;
        const float* pk = xrow + ks * 32 + 8 * g;     // k = ks*32+8g+{0..7}, +16+{0..7}
        float4 a0 = ((const float4*)pk)[0];
        float4 a1 = ((const float4*)pk)[1];
        float4 a2 = ((const float4*)(pk + 16))[0];
        float4 a3 = ((const float4*)(pk + 16))[1];
        A.u[0]  = f2bf(a0.x); A.u[1]  = f2bf(a0.y); A.u[2]  = f2bf(a0.z); A.u[3]  = f2bf(a0.w);
        A.u[4]  = f2bf(a1.x); A.u[5]  = f2bf(a1.y); A.u[6]  = f2bf(a1.z); A.u[7]  = f2bf(a1.w);
        A.u[8]  = f2bf(a2.x); A.u[9]  = f2bf(a2.y); A.u[10] = f2bf(a2.z); A.u[11] = f2bf(a2.w);
        A.u[12] = f2bf(a3.x); A.u[13] = f2bf(a3.y); A.u[14] = f2bf(a3.z); A.u[15] = f2bf(a3.w);
    } else {
        A.q[0] = make_uint4(0, 0, 0, 0);
        A.q[1] = make_uint4(0, 0, 0, 0);
    }
    uint4* out = (uint4*)(afrag + aoff + (size_t)i * 16);
    out[0] = A.q[0];
    out[1] = A.q[1];
}

// ---------------------------------------------------------------- kernel 3
// Windowed q-projection GEMM. One wave computes a 16x64 tile (4 N-tiles):
// per K-step: one A-frag load feeds 4 WMMAs. Pure loads + v_wmma, no branches.
__global__ void __launch_bounds__(256)
qproj_gemm_kernel(const unsigned short* __restrict__ afrag,
                  const unsigned short* __restrict__ qwfrag,
                  const float* __restrict__ qb,
                  float* __restrict__ qws,
                  int level, int Mtiles,
                  unsigned long long aoff, unsigned long long qoff) {
    int tid = threadIdx.x;
    int wid = blockIdx.x * (blockDim.x >> 5) + (tid >> 5);
    if (wid >= Mtiles * 4) return;
    int mtile = wid >> 2;
    int nq    = wid & 3;              // N-tiles nq*4 .. nq*4+3
    int lane  = tid & 31;
    int l15 = lane & 15, g = lane >> 4;

    const unsigned short* abase =
        afrag + aoff + ((size_t)mtile * 256 + lane) * 16;            // +ks*512
    const unsigned short* bbase =
        qwfrag + (size_t)level * 65536 + ((size_t)(nq * 4) * 256 + lane) * 16;  // +j*4096 +ks*512

    v8f acc0 = {0.f,0.f,0.f,0.f,0.f,0.f,0.f,0.f};
    v8f acc1 = acc0, acc2 = acc0, acc3 = acc0;

#pragma unroll
    for (int ks = 0; ks < 8; ks++) {
        BFrag A, B0, B1, B2, B3;
        const uint4* aq = (const uint4*)(abase + ks * 512);
        A.q[0] = aq[0];  A.q[1] = aq[1];
        const uint4* b0 = (const uint4*)(bbase + ks * 512);
        const uint4* b1 = (const uint4*)(bbase + 4096  + ks * 512);
        const uint4* b2 = (const uint4*)(bbase + 8192  + ks * 512);
        const uint4* b3 = (const uint4*)(bbase + 12288 + ks * 512);
        B0.q[0] = b0[0]; B0.q[1] = b0[1];
        B1.q[0] = b1[0]; B1.q[1] = b1[1];
        B2.q[0] = b2[0]; B2.q[1] = b2[1];
        B3.q[0] = b3[0]; B3.q[1] = b3[1];
        acc0 = __builtin_amdgcn_wmma_f32_16x16x32_bf16(false, A.v, false, B0.v, (short)0, acc0, false, false);
        acc1 = __builtin_amdgcn_wmma_f32_16x16x32_bf16(false, A.v, false, B1.v, (short)0, acc1, false, false);
        acc2 = __builtin_amdgcn_wmma_f32_16x16x32_bf16(false, A.v, false, B2.v, (short)0, acc2, false, false);
        acc3 = __builtin_amdgcn_wmma_f32_16x16x32_bf16(false, A.v, false, B3.v, (short)0, acc3, false, false);
    }

    // C/D layout: lane<16 -> N=lane, M=v ; lane>=16 -> N=lane-16, M=8+v
    float* op  = qws + qoff;
    int    mrow = mtile * 16 + g * 8;
    v8f accs[4] = {acc0, acc1, acc2, acc3};
#pragma unroll
    for (int j = 0; j < 4; j++) {
        int   n   = (nq * 4 + j) * 16 + l15;
        float qbv = qb[level * 256 + n];
#pragma unroll
        for (int v = 0; v < 8; v++)
            op[(size_t)(mrow + v) * 256 + n] = accs[j][v] + qbv;
    }
}

// ---------------------------------------------------------------- kernel 4
// One wave32 per (level, b, head, point); lane = offset channel (0..31).
__global__ void __launch_bounds__(256)
stage2_kernel(const float* __restrict__ x,  const float* __restrict__ qws,
              const float* __restrict__ ow0, const float* __restrict__ ow1,
              const float* __restrict__ ow2, const float* __restrict__ ow3,
              const float* __restrict__ ob,  const float* __restrict__ lng,
              const float* __restrict__ lnb, const float* __restrict__ pw,
              float* __restrict__ out) {
    const int Ks[4]  = {7, 5, 3, 1};
    const int STs[4] = {16, 8, 4, 2};
    const int Hs[4]  = {128, 64, 32, 16};
    const int LB[4]  = {0, 16384, 20480, 21504};
    const unsigned long long QO[4] = {0ull, 6422528ull, 9699328ull, 10878976ull};

    int level = blockIdx.y;
    int tid   = threadIdx.x;
    int lane  = tid & 31;
    int grp   = blockIdx.x * (blockDim.x >> 5) + (tid >> 5);  // 0..4095
    if (grp >= 4096) return;
    int b = grp >> 9, rem = grp & 511;
    int h = rem >> 6, p = rem & 63;

    int K = Ks[level], ST = STs[level], H = Hs[level], W = H;
    const float* ow = (level == 0) ? ow0 : (level == 1) ? ow1 : (level == 2) ? ow2 : ow3;

    int o  = lane;
    int py = p >> 3, px = p & 7;
    int g4 = (o >> 2) * 4;    // group of 4 input channels for this output channel

    const float* qbase = qws + QO[level]
                       + ((size_t)(b * 64 + p) * K * K) * 256 + h * 32 + g4;
    float acc = ob[level * 32 + o];
    int kidx = 0;
    for (int ky = 0; ky < K; ky++) {
        int y = py * ST + ky - (K >> 1);
        for (int kx = 0; kx < K; kx++, kidx++) {
            int xc = px * ST + kx - (K >> 1);
            if ((unsigned)y < (unsigned)H && (unsigned)xc < (unsigned)W) {
                const float* qp = qbase + (size_t)kidx * 256;
#pragma unroll
                for (int ic = 0; ic < 4; ic++)
                    acc += ow[((o * 4 + ic) * K + ky) * K + kx] * qp[ic];
            }
        }
    }

    // LayerNorm over 32 channels
    float mu  = wave_sum(acc) * (1.0f / 32.0f);
    float d   = acc - mu;
    float var = wave_sum(d * d) * (1.0f / 32.0f);
    float gn  = d * rsqrtf(var + 1e-5f);
    gn = gn * lng[level * 32 + o] + lnb[level * 32 + o];

    // exact GELU
    float ge = 0.5f * gn * (1.0f + erff(gn * 0.70710678118654752f));

    // 2x32 projection -> (dy, dx)
    float dy = wave_sum(pw[level * 64 + o] * ge);
    float dx = wave_sum(pw[level * 64 + 32 + o] * ge);

    // reference points (8x8 grid) + clip; grid reverses (y,x)->(x,y)
    float ry = (py + 0.5f) * 0.25f - 1.0f;
    float rx = (px + 0.5f) * 0.25f - 1.0f;
    float posy = fminf(fmaxf(dy + ry, -1.0f), 1.0f);
    float posx = fminf(fmaxf(dx + rx, -1.0f), 1.0f);
    float gx = (posx + 1.0f) * 0.5f * (float)(W - 1);
    float gy = (posy + 1.0f) * 0.5f * (float)(H - 1);

    // bilinear sample from ORIGINAL x; lane samples its own channel
    float x0 = floorf(gx), y0 = floorf(gy);
    float wx = gx - x0,    wy = gy - y0;
    int x0i = min(max((int)x0, 0), W - 1);
    int y0i = min(max((int)y0, 0), H - 1);
    int x1i = min(x0i + 1, W - 1);
    int y1i = min(y0i + 1, H - 1);

    const float* img = x + ((size_t)b * LTOT + LB[level]) * EMB + h * 32 + o;
    float v00 = img[(size_t)(y0i * W + x0i) * EMB];
    float v01 = img[(size_t)(y0i * W + x1i) * EMB];
    float v10 = img[(size_t)(y1i * W + x0i) * EMB];
    float v11 = img[(size_t)(y1i * W + x1i) * EMB];
    float v = v00 * (1.f - wx) * (1.f - wy) + v01 * wx * (1.f - wy)
            + v10 * (1.f - wx) * wy        + v11 * wx * wy;

    out[((size_t)b * 256 + (level * 64 + p)) * 256 + h * 32 + o] = v;
}

// ---------------------------------------------------------------- launch

extern "C" void kernel_launch(void* const* d_in, const int* in_sizes, int n_in,
                              void* d_out, int out_size, void* d_ws, size_t ws_size,
                              hipStream_t stream) {
    const float* x   = (const float*)d_in[0];
    const float* qw  = (const float*)d_in[1];
    const float* qb  = (const float*)d_in[2];
    const float* ow0 = (const float*)d_in[3];
    const float* ow1 = (const float*)d_in[4];
    const float* ow2 = (const float*)d_in[5];
    const float* ow3 = (const float*)d_in[6];
    const float* ob  = (const float*)d_in[7];
    const float* lng = (const float*)d_in[8];
    const float* lnb = (const float*)d_in[9];
    const float* pw  = (const float*)d_in[10];
    float* out = (float*)d_out;

    // workspace layout (bytes):
    //   [0,       524288)  bf16 qw B-fragments
    //   [524288,  +22.0MB) bf16 windowed-x A-fragments
    //   [22.5MB,  +44.0MB) f32 q rows
    unsigned short* qwfrag = (unsigned short*)d_ws;
    unsigned short* afrag  = (unsigned short*)((char*)d_ws + 524288);
    float*          qws    = (float*)((char*)d_ws + 524288 + 22020096);

    prep_qw_kernel<<<1024, 256, 0, stream>>>(qw, qwfrag);

    const int Ks[4]  = {7, 5, 3, 1};
    const int STs[4] = {16, 8, 4, 2};
    const int Hs[4]  = {128, 64, 32, 16};
    const int LBs[4] = {0, 16384, 20480, 21504};
    const int Mtiles[4] = {1568, 800, 288, 32};   // rows/16 ; rows = 8*64*K*K
    const unsigned long long QO[4] = {0ull, 6422528ull, 9699328ull, 10878976ull}; // f32 units
    const unsigned long long AO[4] = {0ull, 6422528ull, 9699328ull, 10878976ull}; // u16 units

    for (int lv = 0; lv < 4; lv++) {
        int chunks = Mtiles[lv] * 256;            // 8 ksteps * 32 lanes per M-tile
        aprep_kernel<<<chunks / 256, 256, 0, stream>>>(
            x, afrag, Ks[lv], STs[lv], Hs[lv], LBs[lv], AO[lv], chunks);
    }
    for (int lv = 0; lv < 4; lv++) {
        int waves  = Mtiles[lv] * 4;              // 4 N-quads per M-tile
        int blocks = waves / 8;                   // 8 waves / 256-thread block (exact)
        qproj_gemm_kernel<<<blocks, 256, 0, stream>>>(
            afrag, qwfrag, qb, qws, lv, Mtiles[lv], AO[lv], QO[lv]);
    }

    stage2_kernel<<<dim3(512, 4), 256, 0, stream>>>(
        x, qws, ow0, ow1, ow2, ow3, ob, lng, lnb, pw, out);
}
